// InstrumentedPSI_9062380995374
// MI455X (gfx1250) — matrix operand order repaired and checked by reading, not compiled, not run on hardware
//
#include <hip/hip_runtime.h>
#include <hip/hip_bf16.h>
#include <math.h>

// ---------------------------------------------------------------------------
// Problem constants (match reference): B=4, S=4096, D=1024
// ---------------------------------------------------------------------------
#define PSI_B 4
#define PSI_S 4096
#define PSI_D 1024
#define PSI_M (PSI_B * PSI_S)   // 16384 rows

typedef __attribute__((ext_vector_type(16))) __bf16 v16bf;
typedef __attribute__((ext_vector_type(2)))  __bf16 v2bf;
typedef __attribute__((ext_vector_type(8)))  float  v8f;

union FragBF { v16bf v; unsigned int u[8]; };

__device__ __forceinline__ unsigned int pack2_bf16(float a, float b) {
#if __has_builtin(__builtin_amdgcn_cvt_pk_bf16_f32)
    union { v2bf v; unsigned int u; } cv;
    cv.v = __builtin_amdgcn_cvt_pk_bf16_f32(a, b);
    return cv.u;
#else
    unsigned int ua = __float_as_uint(a), ub = __float_as_uint(b);
    unsigned int ra = (ua + 0x7FFFu + ((ua >> 16) & 1u)) >> 16;
    unsigned int rb = (ub + 0x7FFFu + ((ub >> 16) & 1u)) >> 16;
    return (ra & 0xFFFFu) | (rb << 16);
#endif
}
__device__ __forceinline__ unsigned short f32_to_bf16u(float a) {
    unsigned int ua = __float_as_uint(a);
    return (unsigned short)((ua + 0x7FFFu + ((ua >> 16) & 1u)) >> 16);
}

// ---------------------------------------------------------------------------
// B-operand swizzle: f32 KxN row-major -> bf16 WMMA-B fragment layout.
// uint index q = ((nb*(K/32)+kb)*32 + lane)*8 + v
//   n = nb*16 + (lane&15), K-pair k = kb*32 + (v>>2)*16 + (lane>>4)*8 + (v&3)*2
// (ISA 7.12.2 wave32 16-bit B operand layout)
// ---------------------------------------------------------------------------
__global__ __launch_bounds__(256) void swz_b_kernel(
    const float* __restrict__ W, unsigned int* __restrict__ out, int K, int N) {
    size_t q = (size_t)blockIdx.x * 256 + threadIdx.x;   // K*N/2 uints total
    int v    = (int)(q & 7);
    int lane = (int)((q >> 3) & 31);
    size_t fb = q >> 8;
    int kb = (int)(fb % (size_t)(K >> 5));
    int nb = (int)(fb / (size_t)(K >> 5));
    int k = kb * 32 + ((v >> 2) << 4) + ((lane >> 4) << 3) + ((v & 3) << 1);
    int n = nb * 16 + (lane & 15);
    out[q] = pack2_bf16(W[(size_t)k * N + n], W[(size_t)(k + 1) * N + n]);
}

// ---------------------------------------------------------------------------
// A-operand swizzle: f32 MxK row-major -> bf16 WMMA-A fragment layout.
// uint index q = ((mb*(K/32)+kb)*32 + lane)*8 + v
//   row = mb*16 + (lane&15), K-pair k = kb*32 + (v>>2)*16 + (lane>>4)*8 + (v&3)*2
// (ISA 7.12.2 wave32 16-bit A operand layout)
// ---------------------------------------------------------------------------
__global__ __launch_bounds__(256) void swz_a_kernel(
    const float* __restrict__ A, unsigned int* __restrict__ out, int M, int K) {
    size_t q = (size_t)blockIdx.x * 256 + threadIdx.x;   // M*K/2 uints total
    int v    = (int)(q & 7);
    int lane = (int)((q >> 3) & 31);
    size_t fb = q >> 8;
    int kb = (int)(fb % (size_t)(K >> 5));
    int mb = (int)(fb / (size_t)(K >> 5));
    int row = mb * 16 + (lane & 15);
    int k = kb * 32 + ((v >> 2) << 4) + ((lane >> 4) << 3) + ((v & 3) << 1);
    out[q] = pack2_bf16(A[(size_t)row * K + k], A[(size_t)row * K + k + 1]);
}

// ---------------------------------------------------------------------------
// Unified fragment-direct WMMA GEMM:  C(MxN) = Aswz(bf16) @ Bswz(bf16) + epi
// Both operands pre-swizzled in global memory -> pure register-resident loop:
// no LDS, no barriers; coalesced global_load_b128 -> v_wmma only.
// Block 256 thr = 8 waves (4M x 2N); block tile 128x128; wave tile 32x64
// (2x4 frags, 8 v_wmma / K-stage). Intra-block fragment reuse is served by
// the WGP$; weights / h2 stay resident in the 192MB L2.
// Epilogues: +bias->f32 | +bias,GELU(erf)->bf16 A-frag-swizzled | +bias,+res->f32
// ---------------------------------------------------------------------------
enum EpMode { EPI_BIAS_F32 = 0, EPI_BIAS_GELU_SWZ = 1, EPI_BIAS_RES_F32 = 2 };

template <int EM>
__global__ __launch_bounds__(256) void wmma_gemm_frag_kernel(
    const unsigned int* __restrict__ Asw,     // swizzled bf16 A (MxK)
    const unsigned int* __restrict__ Bsw,     // swizzled bf16 B (KxN)
    const float* __restrict__ bias,           // [N]
    const float* __restrict__ resid,          // [M*N] (residual mode)
    void* __restrict__ out,                   // f32 (MxN) or swizzled bf16
    int M, int N, int K) {
    const int tid  = threadIdx.x;
    const int lane = tid & 31;
    const int wave = tid >> 5;
    const int wm   = wave & 3;
    const int wn   = wave >> 2;
    const int l15  = lane & 15;
    const int grp  = lane >> 4;
    const int mb0  = blockIdx.y * 8;          // 16-row blocks
    const int nb0  = blockIdx.x * 8;          // 16-col blocks
    const int KB   = K >> 5;

    v8f acc[2][4] = {};

    for (int kb = 0; kb < KB; ++kb) {
        FragBF a2[2], b4[4];
#pragma unroll
        for (int i = 0; i < 2; ++i) {
            const unsigned int* p =
                Asw + (((size_t)(mb0 + wm * 2 + i) * KB + kb) * 32 + lane) * 8;
            uint4 lo = *(const uint4*)p;
            uint4 hi = *(const uint4*)(p + 4);
            a2[i].u[0] = lo.x; a2[i].u[1] = lo.y; a2[i].u[2] = lo.z; a2[i].u[3] = lo.w;
            a2[i].u[4] = hi.x; a2[i].u[5] = hi.y; a2[i].u[6] = hi.z; a2[i].u[7] = hi.w;
        }
#pragma unroll
        for (int j = 0; j < 4; ++j) {
            const unsigned int* p =
                Bsw + (((size_t)(nb0 + wn * 4 + j) * KB + kb) * 32 + lane) * 8;
            uint4 lo = *(const uint4*)p;
            uint4 hi = *(const uint4*)(p + 4);
            b4[j].u[0] = lo.x; b4[j].u[1] = lo.y; b4[j].u[2] = lo.z; b4[j].u[3] = lo.w;
            b4[j].u[4] = hi.x; b4[j].u[5] = hi.y; b4[j].u[6] = hi.z; b4[j].u[7] = hi.w;
        }
#pragma unroll
        for (int i = 0; i < 2; ++i)
#pragma unroll
            for (int j = 0; j < 4; ++j)
                acc[i][j] = __builtin_amdgcn_wmma_f32_16x16x32_bf16(
                    false, a2[i].v, false, b4[j].v, (short)0, acc[i][j],
                    false, false);
    }

    // ---- epilogue: C/D layout: row = r + 8*grp, col = l15 ----
#pragma unroll
    for (int i = 0; i < 2; ++i) {
#pragma unroll
        for (int j = 0; j < 4; ++j) {
            int col = (nb0 + wn * 4 + j) * 16 + l15;
            float bv = bias[col];
#pragma unroll
            for (int r = 0; r < 8; ++r) {
                int row = (mb0 + wm * 2 + i) * 16 + r + 8 * grp;
                float v = acc[i][j][r] + bv;
                if constexpr (EM == EPI_BIAS_F32) {
                    ((float*)out)[(size_t)row * N + col] = v;
                } else if constexpr (EM == EPI_BIAS_RES_F32) {
                    size_t oi = (size_t)row * N + col;
                    ((float*)out)[oi] = v + resid[oi];
                } else {
                    // GELU(erf), store into A-fragment swizzled layout
                    // (this output is the next GEMM's A; its K == this N)
                    float g = 0.5f * v * (1.0f + erff(v * 0.70710678118654752f));
                    int half = (col >> 4) & 1;
                    int grp2 = (col >> 3) & 1;
                    int pidx = (col >> 1) & 3;
                    int low  = col & 1;
                    int v2   = half * 4 + pidx;
                    int lane2 = (row & 15) | (grp2 << 4);
                    size_t si =
                        ((((size_t)(row >> 4) * (N >> 5) + (col >> 5)) * 32 +
                          lane2) * 8 + v2) * 2 + low;
                    ((unsigned short*)out)[si] = f32_to_bf16u(g);
                }
            }
        }
    }
}

// ---------------------------------------------------------------------------
// Sequential scan over S: one thread per (b,d) chain; fully coalesced in d.
// Produces ctx (B,S,4D) f32: [cr | ci | rr | ri]
// ---------------------------------------------------------------------------
__global__ __launch_bounds__(256) void scan_kernel(
    const float* __restrict__ x, const float* __restrict__ omega,
    const float* __restrict__ log_scale, float* __restrict__ ctx) {
    int t = blockIdx.x * 256 + threadIdx.x;   // 0 .. B*D-1
    int b = t / PSI_D;
    int d = t % PSI_D;
    float sc = expf(log_scale[d]);
    float phi = 0.0f, sr = 0.0f, si = 0.0f;
    size_t baseIn  = (size_t)b * PSI_S * PSI_D + d;
    size_t baseOut = (size_t)b * PSI_S * 4 * PSI_D + d;
    for (int s = 0; s < PSI_S; ++s) {
        size_t idx = baseIn + (size_t)s * PSI_D;
        float om = omega[idx];
        float xv = x[idx];
        float inv = 1.0f / (float)(s + 1);
        phi += om * sc * rsqrtf((float)(s + 1));
        float sp, cp;
        sincosf(phi, &sp, &cp);
        float cr = xv * cp, ci = xv * sp;
        sr += cr; si += ci;
        float mr = sr * inv, mi = si * inv;
        float rr = mr * cp + mi * sp;
        float ri = mi * cp - mr * sp;
        size_t o = baseOut + (size_t)s * 4 * PSI_D;
        ctx[o]             = cr;
        ctx[o + PSI_D]     = ci;
        ctx[o + 2 * PSI_D] = rr;
        ctx[o + 3 * PSI_D] = ri;
    }
}

// ---------------------------------------------------------------------------
// Fused LayerNorm: per-row stats over F features, then normalize + affine,
// pack to bf16 and write directly in WMMA A-fragment swizzled layout.
// One block per row; row re-read hits WGP$/L2.
// ---------------------------------------------------------------------------
__global__ __launch_bounds__(256) void ln_swz_kernel(
    const float* __restrict__ ctx, const float* __restrict__ gamma,
    const float* __restrict__ beta, unsigned int* __restrict__ a2sw, int F) {
    __shared__ float ssum[256];
    __shared__ float ssq[256];
    __shared__ float smu, srs;
    const int rrow = blockIdx.x;
    const float* row = ctx + (size_t)rrow * F;
    float s = 0.0f, s2 = 0.0f;
    for (int i = threadIdx.x; i < F; i += 256) {
        float v = row[i];
        s += v; s2 += v * v;
    }
    ssum[threadIdx.x] = s; ssq[threadIdx.x] = s2;
    __syncthreads();
    for (int off = 128; off > 0; off >>= 1) {
        if (threadIdx.x < off) {
            ssum[threadIdx.x] += ssum[threadIdx.x + off];
            ssq[threadIdx.x]  += ssq[threadIdx.x + off];
        }
        __syncthreads();
    }
    if (threadIdx.x == 0) {
        float mu  = ssum[0] / (float)F;
        float var = ssq[0] / (float)F - mu * mu;
        smu = mu;
        srs = rsqrtf(var + 1e-5f);
    }
    __syncthreads();
    const float mu = smu, rs = srs;
    const int KB = F >> 5;
    const int mb = rrow >> 4;
    for (int c2 = threadIdx.x; c2 < (F >> 1); c2 += 256) {
        int col = c2 << 1;
        float a0 = (row[col]     - mu) * rs * gamma[col]     + beta[col];
        float a1 = (row[col + 1] - mu) * rs * gamma[col + 1] + beta[col + 1];
        int kb   = col >> 5;
        int half = (col >> 4) & 1;
        int grp2 = (col >> 3) & 1;
        int pidx = (col >> 1) & 3;
        int v2   = half * 4 + pidx;
        int lane2 = (rrow & 15) | (grp2 << 4);
        a2sw[(((size_t)mb * KB + kb) * 32 + lane2) * 8 + v2] = pack2_bf16(a0, a1);
    }
}

// ---------------------------------------------------------------------------
// kernel_launch
// inputs: 0:x 1:W_omega 2:b_omega 3:log_scale 4:ln_gamma 5:ln_beta
//         6:W1 7:b1 8:W2 9:b2     (all f32)
// ---------------------------------------------------------------------------
extern "C" void kernel_launch(void* const* d_in, const int* in_sizes, int n_in,
                              void* d_out, int out_size, void* d_ws, size_t ws_size,
                              hipStream_t stream) {
    (void)in_sizes; (void)n_in; (void)out_size; (void)ws_size;
    const float* x         = (const float*)d_in[0];
    const float* W_omega   = (const float*)d_in[1];
    const float* b_omega   = (const float*)d_in[2];
    const float* log_scale = (const float*)d_in[3];
    const float* ln_gamma  = (const float*)d_in[4];
    const float* ln_beta   = (const float*)d_in[5];
    const float* W1        = (const float*)d_in[6];
    const float* b1        = (const float*)d_in[7];
    const float* W2        = (const float*)d_in[8];
    const float* b2        = (const float*)d_in[9];
    float* out             = (float*)d_out;

    const int M  = PSI_M;          // 16384
    const int D  = PSI_D;          // 1024
    const int F  = 4 * D;          // 4096
    const int H  = 2 * D;          // 2048

    // workspace layout (bytes), lifetime-overlapped:
    //   [0,256M)      ctx (f32)                 live: scan .. ln_swz
    //   [256M,320M)   omega (f32)               live: gemm1 .. scan
    //   [320M,352M)   xsw (bf16 A-frag)         live: swz_a .. gemm1
    //   [256M,384M)   a2sw (bf16 A-frag)        live: ln_swz .. gemm2   (reuses omega+xsw)
    //   [384M,448M)   h2sw (bf16 A-frag)        live: gemm2 .. gemm3
    //   [448M,470M)   wWo / w1b / w2b (bf16 B-frag)  live: all
    char* ws = (char*)d_ws;
    const size_t MB = 1024 * 1024;
    float*        ctx   = (float*)(ws + 0);
    float*        omega = (float*)(ws + 256 * MB);
    unsigned int* xsw   = (unsigned int*)(ws + 320 * MB);
    unsigned int* a2sw  = (unsigned int*)(ws + 256 * MB);
    unsigned int* h2sw  = (unsigned int*)(ws + 384 * MB);
    unsigned int* wWo   = (unsigned int*)(ws + 448 * MB);
    unsigned int* w1b   = (unsigned int*)(ws + 450 * MB);
    unsigned int* w2b   = (unsigned int*)(ws + 466 * MB);

    // 1) weights -> bf16 B-fragment layout; x -> bf16 A-fragment layout
    swz_b_kernel<<<(int)(((size_t)D * D / 2) / 256), 256, 0, stream>>>(W_omega, wWo, D, D);
    swz_b_kernel<<<(int)(((size_t)F * H / 2) / 256), 256, 0, stream>>>(W1, w1b, F, H);
    swz_b_kernel<<<(int)(((size_t)H * D / 2) / 256), 256, 0, stream>>>(W2, w2b, H, D);
    swz_a_kernel<<<(int)(((size_t)M * D / 2) / 256), 256, 0, stream>>>(x, xsw, M, D);

    // 2) omega = x @ W_omega + b_omega        (M=16384, N=1024, K=1024)
    wmma_gemm_frag_kernel<EPI_BIAS_F32><<<dim3(D / 128, M / 128), 256, 0, stream>>>(
        xsw, wWo, b_omega, nullptr, omega, M, D, D);

    // 3) sequential scan -> ctx (B,S,4D)
    scan_kernel<<<(PSI_B * PSI_D) / 256, 256, 0, stream>>>(x, omega, log_scale, ctx);

    // 4) fused LayerNorm + bf16 A-fragment swizzle: ctx -> a2sw
    ln_swz_kernel<<<M, 256, 0, stream>>>(ctx, ln_gamma, ln_beta, a2sw, F);

    // 5) h2sw = swz(gelu(a2sw @ W1 + b1))     (M=16384, N=2048, K=4096)
    wmma_gemm_frag_kernel<EPI_BIAS_GELU_SWZ><<<dim3(H / 128, M / 128), 256, 0, stream>>>(
        a2sw, w1b, b1, nullptr, h2sw, M, H, F);

    // 6) out = x + h2sw @ W2 + b2             (M=16384, N=1024, K=2048)
    wmma_gemm_frag_kernel<EPI_BIAS_RES_F32><<<dim3(D / 128, M / 128), 256, 0, stream>>>(
        h2sw, w2b, b2, x, out, M, D, H);
}
